// AnoNAViLa_Model_51178830299519
// MI455X (gfx1250) — compile-verified, matrix operand order checked by reading, not run on hardware
//
#include <hip/hip_runtime.h>
#include <hip/hip_bf16.h>
#include <stdint.h>

// ---------------------------------------------------------------------------
// Problem constants (AnoNAViLa): B=1024, N=96 per branch, D=512.
// Factorized: h1[b,n,:] = relu(A[b,:] + e[b,n]*T[n,:]) with
//   A = img@W1[:512,:] + b1   (1024x512)
//   T = text@W1[512:,:]       (96x512 per branch)
//   e[b,n] = exp(softmax(scale * img@text^T)[b,n])
// Then h2 = relu(h1@W2 + b2), h3 = h2@W3 (+b3 folded post-mean), out = mean_n h3.
// ---------------------------------------------------------------------------

#define BB 1024
#define NN 96
#define DD 512

typedef __attribute__((ext_vector_type(16))) __bf16 v16bf;
typedef __attribute__((ext_vector_type(8)))  float  v8f;

union FragU {
    uint32_t u[8];
    uint4    q[2];
    v16bf    b;
};

__device__ __forceinline__ unsigned short f2bf(float f) {
    uint32_t u = __float_as_uint(f);
    u += 0x7FFFu + ((u >> 16) & 1u);   // round-to-nearest-even
    return (unsigned short)(u >> 16);
}
__device__ __forceinline__ uint32_t pack_bf2(float lo, float hi) {
    return (uint32_t)f2bf(lo) | ((uint32_t)f2bf(hi) << 16);
}

// Fragment layout (16-bit A-matrix 16x32 per ISA 7.12.2; B uses same index map
// with N in place of M). Element (m, kk): lane = m + 16*((kk%16)>=8),
// r = (kk/16)*4 + (kk%8)/2, half = kk%2. Inverse used below.
__device__ __forceinline__ int frag_kk(int lane, int r) {
    // even kk for half=0; +1 for half=1
    return ((r >> 2) << 4) + ((lane >> 4) << 3) + ((r & 3) << 1);
}

// ---------------------------------------------------------------------------
// Kernel 1: sims -> softmax -> e = exp(softmax weight).  grid (B, 2), 128 thr.
// ---------------------------------------------------------------------------
__global__ __launch_bounds__(128) void k_softmax_e(
    const float* __restrict__ img, const float* __restrict__ tN,
    const float* __restrict__ tA, const float* __restrict__ logit,
    float* __restrict__ eB)
{
    int b  = blockIdx.x;
    int br = blockIdx.y;
    const float* text = (br == 0) ? tN : tA;
    __shared__ float red[128];
    int n = threadIdx.x;
    float scale = __expf(logit[0]);

    float sims = -3.0e38f;
    if (n < NN) {
        const float* ir = img + b * DD;
        const float* tr = text + n * DD;
        float acc = 0.f;
        #pragma unroll 8
        for (int j = 0; j < DD; ++j) acc += ir[j] * tr[j];
        sims = acc * scale;
    }
    red[n] = sims;
    __syncthreads();
    #pragma unroll
    for (int off = 64; off > 0; off >>= 1) {
        if (n < off) red[n] = fmaxf(red[n], red[n + off]);
        __syncthreads();
    }
    float mx = red[0];
    __syncthreads();
    float p = (n < NN) ? __expf(sims - mx) : 0.f;
    red[n] = p;
    __syncthreads();
    #pragma unroll
    for (int off = 64; off > 0; off >>= 1) {
        if (n < off) red[n] += red[n + off];
        __syncthreads();
    }
    float ssum = red[0];
    if (n < NN) eB[(br * BB + b) * NN + n] = __expf(p / ssum);
}

// ---------------------------------------------------------------------------
// Kernel 2: A = img@W1_top + b1 (rows 0..1023); T rows (1024..1215).
// grid 1216 blocks x 256 threads; each thread -> 2 output columns.
// ---------------------------------------------------------------------------
__global__ __launch_bounds__(256) void k_prep_at(
    const float* __restrict__ img, const float* __restrict__ tN,
    const float* __restrict__ tA, const float* __restrict__ W1,
    const float* __restrict__ b1, float* __restrict__ Ab,
    float* __restrict__ Tb)
{
    int row = blockIdx.x;
    int tid = threadIdx.x;
    __shared__ float x[DD];

    const float* src; const float* wbase; float* dst; bool addb = false;
    if (row < BB)            { src = img + row * DD;          wbase = W1;            dst = Ab + row * DD;           addb = true; }
    else if (row < BB + NN)  { int n = row - BB;  src = tN + n * DD; wbase = W1 + DD * DD; dst = Tb + n * DD; }
    else                     { int n = row - BB - NN; src = tA + n * DD; wbase = W1 + DD * DD; dst = Tb + (NN + n) * DD; }

    x[tid]       = src[tid];
    x[tid + 256] = src[tid + 256];
    __syncthreads();

    int c0 = tid, c1 = tid + 256;
    float a0 = addb ? b1[c0] : 0.f;
    float a1 = addb ? b1[c1] : 0.f;
    #pragma unroll 4
    for (int j = 0; j < DD; ++j) {
        float xv = x[j];
        const float* wr = wbase + j * DD;
        a0 += xv * wr[c0];
        a1 += xv * wr[c1];
    }
    dst[c0] = a0;
    dst[c1] = a1;
}

// ---------------------------------------------------------------------------
// Kernel 3: swizzle W2 (512x256) / W3 (256x128) f32 -> bf16 B-fragment order.
// Fragment = 256 dwords: dword index = lane*8 + r; dword = {K=kk, K=kk+1}.
// W2 frags: (kt 0..15, nt 0..15) -> 65536 dwords. W3: (kt 0..7, nt 0..7).
// ---------------------------------------------------------------------------
__global__ __launch_bounds__(256) void k_conv_w(
    const float* __restrict__ W2, const float* __restrict__ W3,
    uint32_t* __restrict__ w2f, uint32_t* __restrict__ w3f)
{
    int d = blockIdx.x * 256 + threadIdx.x;   // 81920 total
    if (d < 65536) {
        int frag = d >> 8, kt = frag >> 4, nt = frag & 15;
        int lane = (d >> 3) & 31, r = d & 7;
        int n  = lane & 15;
        int kk = frag_kk(lane, r);
        int k  = (kt << 5) + kk;
        int col = (nt << 4) + n;
        w2f[d] = pack_bf2(W2[k * 256 + col], W2[(k + 1) * 256 + col]);
    } else {
        int d2 = d - 65536;
        int frag = d2 >> 8, kt = frag >> 3, nt = frag & 7;
        int lane = (d2 >> 3) & 31, r = d2 & 7;
        int n  = lane & 15;
        int kk = frag_kk(lane, r);
        int k  = (kt << 5) + kk;
        int col = (nt << 4) + n;
        w3f[d2] = pack_bf2(W3[k * 128 + col], W3[(k + 1) * 128 + col]);
    }
}

// ---------------------------------------------------------------------------
// Kernel 4: fused MLP + mean.  grid (B, 2), 256 threads = 8 waves.
// Per chunk of 16 n-rows:
//   h1 tile (16x512 bf16) built in LDS in A-fragment layout
//   GEMM1: h1 @ W2 -> [16,256]; wave w owns N-tiles 2w, 2w+1 (v_wmma bf16)
//   relu(+b2) -> h2 in LDS A-fragment layout (16x256)
//   GEMM2: h2 @ W3 -> [16,128]; wave w owns N-tile w; accumulates across chunks
// Finalize: reduce M inside accumulator + xor-16 shuffle, /96, +b3.
// ---------------------------------------------------------------------------
__global__ __launch_bounds__(256) void k_fused_mlp(
    const float* __restrict__ Ab, const float* __restrict__ Tb,
    const float* __restrict__ eB, const float* __restrict__ b2,
    const float* __restrict__ b3, const uint32_t* __restrict__ w2f,
    const uint32_t* __restrict__ w3f, float* __restrict__ out)
{
    int b  = blockIdx.x;
    int br = blockIdx.y;
    const float* Arow = Ab + b * DD;
    const float* Tbr  = Tb + br * NN * DD;
    const float* erow = eB + (br * BB + b) * NN;

    __shared__ uint32_t h1s[16 * 32 * 8];   // 16 frags  (K=512) : 16 KB
    __shared__ uint32_t h2s[8 * 32 * 8];    //  8 frags  (K=256) :  8 KB

    int tid  = threadIdx.x;
    int lane = tid & 31;
    int w    = tid >> 5;
    int nt0  = w * 2, nt1 = w * 2 + 1;

    v8f acc3 = {};   // persistent h3 accumulator (sum over all 96 rows)

    for (int chunk = 0; chunk < 6; ++chunk) {
        int n0 = chunk * 16;

        // ---- build h1 tile straight into A-fragment layout -----------------
        #pragma unroll
        for (int i = 0; i < 16; ++i) {
            int d    = tid + i * 256;
            int frag = d >> 8;
            int ln   = (d >> 3) & 31;
            int r    = d & 7;
            int m    = ln & 15;
            int k    = (frag << 5) + frag_kk(ln, r);
            float ev = erow[n0 + m];
            const float* trow = Tbr + (n0 + m) * DD;
            float v0 = Arow[k]     + ev * trow[k];
            float v1 = Arow[k + 1] + ev * trow[k + 1];
            v0 = v0 > 0.f ? v0 : 0.f;
            v1 = v1 > 0.f ? v1 : 0.f;
            h1s[d] = pack_bf2(v0, v1);
        }
        __syncthreads();

        // ---- GEMM1: h1 @ W2, two N-tiles per wave --------------------------
        v8f c0 = {}; v8f c1 = {};
        for (int kt = 0; kt < 16; ++kt) {
            FragU a;
            const uint32_t* ap = &h1s[(kt << 8) + (lane << 3)];
            #pragma unroll
            for (int r = 0; r < 8; ++r) a.u[r] = ap[r];

            FragU b0f, b1f;
            const uint4* g0 = reinterpret_cast<const uint4*>(
                w2f + ((((kt << 4) + nt0) << 8) + (lane << 3)));
            const uint4* g1 = reinterpret_cast<const uint4*>(
                w2f + ((((kt << 4) + nt1) << 8) + (lane << 3)));
            b0f.q[0] = g0[0]; b0f.q[1] = g0[1];
            b1f.q[0] = g1[0]; b1f.q[1] = g1[1];

            c0 = __builtin_amdgcn_wmma_f32_16x16x32_bf16(
                     false, a.b, false, b0f.b, (short)0, c0, false, false);
            c1 = __builtin_amdgcn_wmma_f32_16x16x32_bf16(
                     false, a.b, false, b1f.b, (short)0, c1, false, false);
        }

        // ---- relu(+b2) -> h2 LDS (A-fragment layout, K dim = 256) ----------
        unsigned short* h2h = reinterpret_cast<unsigned short*>(h2s);
        #pragma unroll
        for (int t = 0; t < 2; ++t) {
            int nt = t ? nt1 : nt0;
            #pragma unroll
            for (int j = 0; j < 8; ++j) {
                float v = (t ? c1[j] : c0[j]);
                int m   = j + ((lane >> 4) << 3);
                int n   = (nt << 4) + (lane & 15);
                v += b2[n];
                v = v > 0.f ? v : 0.f;
                int kk   = n & 31;
                int frag = n >> 5;
                int ln2  = m + ((((kk & 15) >= 8) ? 1 : 0) << 4);
                int r2   = ((kk >> 4) << 2) + ((kk & 7) >> 1);
                h2h[(((frag << 8) + (ln2 << 3) + r2) << 1) + (kk & 1)] = f2bf(v);
            }
        }
        __syncthreads();

        // ---- GEMM2: h2 @ W3, one N-tile per wave, accumulate ---------------
        for (int kt = 0; kt < 8; ++kt) {
            FragU a;
            const uint32_t* ap = &h2s[(kt << 8) + (lane << 3)];
            #pragma unroll
            for (int r = 0; r < 8; ++r) a.u[r] = ap[r];

            FragU bf;
            const uint4* g = reinterpret_cast<const uint4*>(
                w3f + ((((kt << 3) + w) << 8) + (lane << 3)));
            bf.q[0] = g[0]; bf.q[1] = g[1];

            acc3 = __builtin_amdgcn_wmma_f32_16x16x32_bf16(
                       false, a.b, false, bf.b, (short)0, acc3, false, false);
        }
        __syncthreads();
    }

    // ---- mean over 96 rows: sum 8 M-rows in VGPRs, fold lane-halves --------
    float s = 0.f;
    #pragma unroll
    for (int j = 0; j < 8; ++j) s += acc3[j];
    s += __shfl_xor(s, 16, 32);
    if (lane < 16) {
        int col = (w << 4) + lane;
        out[(br * BB + b) * 128 + col] = s * (1.0f / 96.0f) + b3[col];
    }
}

// ---------------------------------------------------------------------------
// Launch
// ---------------------------------------------------------------------------
extern "C" void kernel_launch(void* const* d_in, const int* in_sizes, int n_in,
                              void* d_out, int out_size, void* d_ws, size_t ws_size,
                              hipStream_t stream) {
    const float* img   = (const float*)d_in[0];   // [1024,512]
    const float* tN    = (const float*)d_in[1];   // [96,512]
    const float* tA    = (const float*)d_in[2];   // [96,512]
    const float* logit = (const float*)d_in[3];   // [1]
    const float* W1    = (const float*)d_in[4];   // [1024,512]
    const float* b1    = (const float*)d_in[5];   // [512]
    const float* W2    = (const float*)d_in[6];   // [512,256]
    const float* b2    = (const float*)d_in[7];   // [256]
    const float* W3    = (const float*)d_in[8];   // [256,128]
    const float* b3    = (const float*)d_in[9];   // [128]
    float* out = (float*)d_out;                   // [2][1024][128]

    char* ws = (char*)d_ws;
    float*    eB  = (float*)(ws);                         // 2*1024*96  f32 = 768 KB
    float*    Ab  = (float*)(ws + 0x000C0000);            // 1024*512   f32 = 2 MB
    float*    Tb  = (float*)(ws + 0x002C0000);            // 2*96*512   f32 = 384 KB
    uint32_t* w2f = (uint32_t*)(ws + 0x00320000);         // 65536 dwords = 256 KB
    uint32_t* w3f = (uint32_t*)(ws + 0x00360000);         // 16384 dwords = 64 KB

    k_softmax_e<<<dim3(BB, 2), 128, 0, stream>>>(img, tN, tA, logit, eB);
    k_prep_at  <<<BB + 2 * NN, 256, 0, stream>>>(img, tN, tA, W1, b1, Ab, Tb);
    k_conv_w   <<<(65536 + 16384) / 256, 256, 0, stream>>>(W2, W3, w2f, w3f);
    k_fused_mlp<<<dim3(BB, 2), 256, 0, stream>>>(Ab, Tb, eB, b2, b3, w2f, w3f, out);
}
